// TrajectronPlusPlus_19670950216143
// MI455X (gfx1250) — matrix operand order repaired and checked by reading, not compiled, not run on hardware
//
#include <hip/hip_runtime.h>
#include <hip/hip_bf16.h>

// ---------------- constants (match reference) ----------------
#define NPED   8192
#define OBSL   8
#define PREDL  12
#define DIM    128
#define EMBD   64
#define SSZ    32
#define MODES  5

typedef __attribute__((ext_vector_type(16))) _Float16 v16h;
typedef __attribute__((ext_vector_type(8)))  float    v8f;

// ---------------- f16 weight/activation pool layout (element offsets) ----------------
static const size_t O_WIH  = 0;                          // 512x64   (swizzled tiles)
static const size_t O_WHH  = O_WIH  + 512*64;            // 512x128
static const size_t O_RPW2 = O_WHH  + 512*128;           // 128x64
static const size_t O_WK   = O_RPW2 + 128*64;            // 128x128
static const size_t O_WV   = O_WK   + 128*128;           // 128x128
static const size_t O_WQ   = O_WV   + 128*128;           // 128x128
static const size_t O_WO   = O_WQ   + 128*128;           // 128x128
static const size_t O_OPW  = O_WO   + 128*128;           // 128x256
static const size_t O_MPW1 = O_OPW  + 128*256;           // 128x256
static const size_t O_TDW1 = O_MPW1 + 128*256;           // 5x128x256 (640x256)
static const size_t O_TDW2 = O_TDW1 + 5*128*256;         // 5x128x128 (640x128)
static const size_t O_TDW3 = O_TDW2 + 5*128*128;         // 5x32x128 (rows 24..31 zero)
static const size_t O_CTX  = O_TDW3 + 5*32*128;          // 8192x128 f16 attention context
static const size_t O_FALL = O_CTX  + (size_t)NPED*128;  // 8192x256 f16 f_all
static const size_t F16_TOTAL = O_FALL + (size_t)NPED*256;

// ---------------- WMMA fragment helpers (gfx1250 wave32 layouts) ----------------
__device__ __forceinline__ float sigm(float x) { return 1.f / (1.f + __expf(-x)); }

// A: 16x32 f16, row-major source with leading dim lda (src pre-offset by k0)
__device__ __forceinline__ v16h frag_A(const _Float16* src, int lda) {
  int lane = threadIdx.x & 31;
  int m  = lane & 15;
  int kb = (lane >> 4) << 4;            // 0 or 16
  const _Float16* row = src + m * lda + kb;
  v16h a;
#pragma unroll
  for (int e = 0; e < 8; ++e) { a[e] = row[e]; a[e + 8] = row[8 + e]; }
  return a;
}

// B: 32x16 tile of W^T, read from PRE-SWIZZLED fragment-major storage:
// tile (nt,kt) lives at ((nt*ktiles)+kt)*512 elems; lane L owns elems [L*16, L*16+16).
// Interface kept as (W, in_dim, n0, k0) so call sites are unchanged.
__device__ __forceinline__ v16h frag_B(const _Float16* Wsw, int in_dim, int n0, int k0) {
  int ktiles = in_dim >> 5;
  const _Float16* p = Wsw + ((size_t)((n0 >> 4) * ktiles + (k0 >> 5)) << 9)
                          + ((threadIdx.x & 31) << 4);
  v16h b;
#pragma unroll
  for (int e = 0; e < 16; ++e) b[e] = p[e];
  return b;
}

// D/C: 16x16 f32; lane holds column n=lane%16, rows (lane>=16?8:0)+r
__device__ __forceinline__ void store_D(float* dst, int ldd, v8f d) {
  int lane = threadIdx.x & 31;
  int n  = lane & 15;
  int mb = (lane >> 4) << 3;
#pragma unroll
  for (int r = 0; r < 8; ++r) dst[(size_t)(mb + r) * ldd + n] = d[r];
}

__device__ __forceinline__ v8f wmma16(v16h a, v16h b, v8f c) {
  return __builtin_amdgcn_wmma_f32_16x16x32_f16(false, a, false, b, (short)0, c, false, false);
}

__device__ __forceinline__ v8f bcast8(float v) {
  v8f c;
#pragma unroll
  for (int r = 0; r < 8; ++r) c[r] = v;
  return c;
}

// ---------------- prep: f32 weights -> f16 fragment-major (swizzled) ----------------
// dst element i: tile = i/512, lane = (i/16)%32, e = i%16
//   nt = tile/ktiles, kt = tile%ktiles
//   row = nt*16 + (lane&15) ; k = kt*32 + ((lane>>4)<<4) + e
__global__ void k_swz(const float* __restrict__ src, _Float16* __restrict__ dst,
                      int out_dim, int in_dim) {
  int i = blockIdx.x * 256 + threadIdx.x;
  int total = out_dim * in_dim;
  if (i >= total) return;
  int ktiles = in_dim >> 5;
  int e = i & 15;
  int lane = (i >> 4) & 31;
  int tile = i >> 9;
  int nt = tile / ktiles, kt = tile - nt * ktiles;
  int row = nt * 16 + (lane & 15);
  int k = kt * 32 + ((lane >> 4) << 4) + e;
  dst[i] = (_Float16)src[(size_t)row * in_dim + k];
}

// td_w3: (5,24,128) padded per-mode to 32 rows (rows 24..31 zero), then swizzled
__global__ void k_swz_tdw3(const float* __restrict__ src, _Float16* __restrict__ dst) {
  int i = blockIdx.x * 256 + threadIdx.x;
  const int total = MODES * 32 * 128, ktiles = 4;
  if (i >= total) return;
  int e = i & 15;
  int lane = (i >> 4) & 31;
  int tile = i >> 9;
  int nt = tile / ktiles, kt = tile - nt * ktiles;
  int row = nt * 16 + (lane & 15);       // 0..159 across 5 modes of 32 rows
  int k = kt * 32 + ((lane >> 4) << 4) + e;
  int m = row >> 5, r = row & 31;
  dst[i] = (r < PREDL * 2) ? (_Float16)src[((size_t)m * 24 + r) * 128 + k] : (_Float16)0.0f;
}

// ---------------- LSTM: 16 peds per wave, gates via WMMA ----------------
__global__ __launch_bounds__(32)
void k_lstm(const float* __restrict__ obs, const float* __restrict__ pe_w,
            const float* __restrict__ pe_b, const float* __restrict__ bih,
            const float* __restrict__ bhh, const _Float16* __restrict__ wih16,
            const _Float16* __restrict__ whh16, float* __restrict__ nf32,
            _Float16* __restrict__ fall16) {
  __shared__ _Float16 sX[16 * 64];
  __shared__ _Float16 sH[16 * 128];
  __shared__ float    sC[16 * 128];
  __shared__ float    sG[16 * 512];
  const int lane = threadIdx.x;
  const int p0 = blockIdx.x * 16;

  // warm L2 with the recurrent weight stream (global_prefetch_b8)
  if (blockIdx.x == 0) {
    const char* w0 = (const char*)wih16;   // wih16/whh16 contiguous in pool
    const size_t bytes = (512 * 64 + 512 * 128) * sizeof(_Float16);
    for (size_t off = (size_t)lane * 256; off < bytes; off += 32 * 256)
      __builtin_prefetch(w0 + off, 0, 1);
  }

  for (int i = lane; i < 16 * 128; i += 32) { sH[i] = (_Float16)0.f; sC[i] = 0.f; }
  __syncthreads();

  for (int t = 0; t < OBSL; ++t) {
    // position embedding for this timestep
    for (int i = lane; i < 16 * 64; i += 32) {
      int p = i >> 6, e = i & 63;
      float ox = obs[(size_t)t * NPED * 2 + (size_t)(p0 + p) * 2 + 0];
      float oy = obs[(size_t)t * NPED * 2 + (size_t)(p0 + p) * 2 + 1];
      sX[i] = (_Float16)(ox * pe_w[e * 2 + 0] + oy * pe_w[e * 2 + 1] + pe_b[e]);
    }
    __syncthreads();
    // gates = X@Wih^T + H@Whh^T + biases : 32 column tiles of 16
    for (int tile = 0; tile < 32; ++tile) {
      int g0 = tile * 16;
      v8f acc = bcast8(bih[g0 + (lane & 15)] + bhh[g0 + (lane & 15)]);
#pragma unroll
      for (int k0 = 0; k0 < 64; k0 += 32)
        acc = wmma16(frag_A(sX + k0, 64), frag_B(wih16, 64, g0, k0), acc);
#pragma unroll
      for (int k0 = 0; k0 < 128; k0 += 32)
        acc = wmma16(frag_A(sH + k0, 128), frag_B(whh16, 128, g0, k0), acc);
      store_D(sG + g0, 512, acc);
    }
    __syncthreads();
    // pointwise LSTM cell update
    for (int i = lane; i < 16 * 128; i += 32) {
      int p = i >> 7, d = i & 127;
      float gi = sG[p * 512 + d];
      float gf = sG[p * 512 + 128 + d];
      float gg = sG[p * 512 + 256 + d];
      float go = sG[p * 512 + 384 + d];
      float c = sigm(gf) * sC[i] + sigm(gi) * tanhf(gg);
      float h = sigm(go) * tanhf(c);
      sC[i] = c;
      sH[i] = (_Float16)h;
      if (t == OBSL - 1) {
        nf32[(size_t)(p0 + p) * 128 + d] = h;
        fall16[(size_t)(p0 + p) * 256 + d] = (_Float16)h;
      }
    }
    __syncthreads();
  }
}

// ---------------- q projection: q = feat@wq^T + bq (WMMA, 16 peds/wave) ----------------
__global__ __launch_bounds__(32)
void k_qproj(const _Float16* __restrict__ fall16, const _Float16* __restrict__ wq16,
             const float* __restrict__ bq, float* __restrict__ q32) {
  __shared__ _Float16 sFeat[16 * 128];
  __shared__ float    sQ[16 * 128];
  const int lane = threadIdx.x;
  const int p0 = blockIdx.x * 16;
  for (int idx = lane; idx < 16 * 128; idx += 32)
    sFeat[idx] = fall16[(size_t)(p0 + (idx >> 7)) * 256 + (idx & 127)];
  __syncthreads();
  for (int nt = 0; nt < 8; ++nt) {
    int n0 = nt * 16;
    v8f acc = bcast8(bq[n0 + (lane & 15)]);
#pragma unroll
    for (int k0 = 0; k0 < 128; k0 += 32)
      acc = wmma16(frag_A(sFeat + k0, 128), frag_B(wq16, 128, n0, k0), acc);
    store_D(sQ + n0, 128, acc);
  }
  __syncthreads();
  for (int idx = lane; idx < 16 * 128; idx += 32)
    q32[(size_t)p0 * 128 + idx] = sQ[idx];
}

// ---------------- attention core: one wave per (scene, query ped) ----------------
__global__ __launch_bounds__(32)
void k_attn(const float* __restrict__ obs, const float* __restrict__ rp_w1,
            const float* __restrict__ rp_b1, const _Float16* __restrict__ rpw2_16,
            const float* __restrict__ rp_b2, const _Float16* __restrict__ wk16,
            const float* __restrict__ bk, const _Float16* __restrict__ wv16,
            const float* __restrict__ bv, const float* __restrict__ q32,
            const float* __restrict__ nf32, _Float16* __restrict__ ctx16) {
  __shared__ _Float16 sR1[32 * 64];
  __shared__ _Float16 sKV[32 * 128];
  __shared__ float sKP[32 * 128];
  __shared__ float sVP[32 * 128];
  __shared__ float sQ[128];
  __shared__ float sAt[128];
  const int lane = threadIdx.x;
  const int b = blockIdx.x >> 5;
  const int i = blockIdx.x & 31;
  const int sp = b * SSZ;
  const float* lp = obs + (size_t)(OBSL - 1) * NPED * 2;  // last positions
  const float pix = lp[(size_t)(sp + i) * 2 + 0];
  const float piy = lp[(size_t)(sp + i) * 2 + 1];

  // relative-position MLP layer 1 (relu), rel[i,j] = pos[j]-pos[i]
  for (int idx = lane; idx < 32 * 64; idx += 32) {
    int j = idx >> 6, h = idx & 63;
    float dx = lp[(size_t)(sp + j) * 2 + 0] - pix;
    float dy = lp[(size_t)(sp + j) * 2 + 1] - piy;
    float v = dx * rp_w1[h * 2 + 0] + dy * rp_w1[h * 2 + 1] + rp_b1[h];
    sR1[idx] = (_Float16)fmaxf(v, 0.f);
  }
  __syncthreads();
  // layer 2 (64->128) via WMMA, result staged in sKP
  for (int mt = 0; mt < 2; ++mt) {
    int j0 = mt * 16;
    for (int nt = 0; nt < 8; ++nt) {
      int n0 = nt * 16;
      v8f acc = bcast8(rp_b2[n0 + (lane & 15)]);
#pragma unroll
      for (int k0 = 0; k0 < 64; k0 += 32)
        acc = wmma16(frag_A(sR1 + j0 * 64 + k0, 64), frag_B(rpw2_16, 64, n0, k0), acc);
      store_D(sKP + j0 * 128 + n0, 128, acc);
    }
  }
  __syncthreads();
  // kv = feat[j] + rel_enc  -> f16
  for (int idx = lane; idx < 32 * 128; idx += 32) {
    int j = idx >> 7, d = idx & 127;
    sKV[idx] = (_Float16)(sKP[idx] + nf32[(size_t)(sp + j) * 128 + d]);
  }
  __syncthreads();
  // kp = kv@wk^T + bk ; vp = kv@wv^T + bv  (WMMA)
  for (int mt = 0; mt < 2; ++mt) {
    int j0 = mt * 16;
    for (int nt = 0; nt < 8; ++nt) {
      int n0 = nt * 16;
      v8f ak = bcast8(bk[n0 + (lane & 15)]);
      v8f av = bcast8(bv[n0 + (lane & 15)]);
#pragma unroll
      for (int k0 = 0; k0 < 128; k0 += 32) {
        v16h a = frag_A(sKV + j0 * 128 + k0, 128);
        ak = wmma16(a, frag_B(wk16, 128, n0, k0), ak);
        av = wmma16(a, frag_B(wv16, 128, n0, k0), av);
      }
      store_D(sKP + j0 * 128 + n0, 128, ak);
      store_D(sVP + j0 * 128 + n0, 128, av);
    }
  }
  // fetch q row for this i (precomputed by k_qproj)
  for (int r = 0; r < 4; ++r) {
    int d = lane + 32 * r;
    sQ[d] = q32[(size_t)(sp + i) * 128 + d];
  }
  __syncthreads();
  // scores (h=4 heads, j=32)
  for (int r = 0; r < 4; ++r) {
    int id = lane + 32 * r;
    int h = id >> 5, j = id & 31;
    float s = 0.f;
    for (int d = 0; d < 32; ++d) s += sQ[h * 32 + d] * sKP[j * 128 + h * 32 + d];
    sAt[id] = s * 0.17677669529663687f;  // 1/sqrt(32)
  }
  __syncthreads();
  if (lane < 4) {
    float mx = -1e30f;
    for (int j = 0; j < 32; ++j) mx = fmaxf(mx, sAt[lane * 32 + j]);
    float sm = 0.f;
    for (int j = 0; j < 32; ++j) { float e = __expf(sAt[lane * 32 + j] - mx); sAt[lane * 32 + j] = e; sm += e; }
    float inv = 1.f / sm;
    for (int j = 0; j < 32; ++j) sAt[lane * 32 + j] *= inv;
  }
  __syncthreads();
  // ctx -> f16 buffer (consumed by k_edge WMMA)
  for (int r = 0; r < 4; ++r) {
    int id = lane + 32 * r;
    int h = id >> 5, d = id & 31;
    float c = 0.f;
    for (int j = 0; j < 32; ++j) c += sAt[h * 32 + j] * sVP[j * 128 + h * 32 + d];
    ctx16[(size_t)(sp + i) * 128 + id] = (_Float16)c;
  }
}

// ---------------- edge head: attn_out + edge projection, all WMMA ----------------
__global__ __launch_bounds__(32)
void k_edge(const _Float16* __restrict__ fall16, const _Float16* __restrict__ ctx16,
            const _Float16* __restrict__ wo16, const float* __restrict__ bo,
            const _Float16* __restrict__ opw16, const float* __restrict__ op_b,
            _Float16* __restrict__ fall16_out) {
  __shared__ _Float16 sCtx[16 * 128];
  __shared__ _Float16 sFeat[16 * 128];
  __shared__ float    sAO[16 * 128];
  __shared__ _Float16 sAOh[16 * 128];
  __shared__ float    sE[16 * 128];
  const int lane = threadIdx.x;
  const int p0 = blockIdx.x * 16;
  for (int idx = lane; idx < 16 * 128; idx += 32) {
    sCtx[idx] = ctx16[(size_t)p0 * 128 + idx];
    sFeat[idx] = fall16[(size_t)(p0 + (idx >> 7)) * 256 + (idx & 127)];
  }
  __syncthreads();
  // attn_out = ctx@wo^T + bo
  for (int nt = 0; nt < 8; ++nt) {
    int n0 = nt * 16;
    v8f acc = bcast8(bo[n0 + (lane & 15)]);
#pragma unroll
    for (int k0 = 0; k0 < 128; k0 += 32)
      acc = wmma16(frag_A(sCtx + k0, 128), frag_B(wo16, 128, n0, k0), acc);
    store_D(sAO + n0, 128, acc);
  }
  __syncthreads();
  for (int idx = lane; idx < 16 * 128; idx += 32) sAOh[idx] = (_Float16)sAO[idx];
  __syncthreads();
  // edge = feat@op_w[:, :128]^T + attn_out@op_w[:, 128:]^T + op_b
  for (int nt = 0; nt < 8; ++nt) {
    int n0 = nt * 16;
    v8f acc = bcast8(op_b[n0 + (lane & 15)]);
#pragma unroll
    for (int k0 = 0; k0 < 128; k0 += 32)
      acc = wmma16(frag_A(sFeat + k0, 128), frag_B(opw16, 256, n0, k0), acc);
#pragma unroll
    for (int k0 = 0; k0 < 128; k0 += 32)
      acc = wmma16(frag_A(sAOh + k0, 128), frag_B(opw16, 256, n0, 128 + k0), acc);
    store_D(sE + n0, 128, acc);
  }
  __syncthreads();
  for (int idx = lane; idx < 16 * 128; idx += 32)
    fall16_out[(size_t)(p0 + (idx >> 7)) * 256 + 128 + (idx & 127)] = (_Float16)sE[idx];
}

// ---------------- decoders: mode probs + 5-mode trajectory heads ----------------
__global__ __launch_bounds__(32)
void k_dec(const _Float16* __restrict__ fall16, const _Float16* __restrict__ mpw1_16,
           const float* __restrict__ mp_b1, const float* __restrict__ mp_w2,
           const float* __restrict__ mp_b2, const _Float16* __restrict__ tdw1_16,
           const float* __restrict__ td_b1, const _Float16* __restrict__ tdw2_16,
           const float* __restrict__ td_b2, const _Float16* __restrict__ tdw3_16,
           const float* __restrict__ td_b3, const float* __restrict__ obs,
           float* __restrict__ out) {
  __shared__ _Float16 sF[16 * 256];
  __shared__ float    sH1[16 * 128];
  __shared__ _Float16 sHh[16 * 128];
  __shared__ float    sT[16 * 32];
  const int lane = threadIdx.x;
  const int p0 = blockIdx.x * 16;

  for (int idx = lane; idx < 16 * 256; idx += 32) sF[idx] = fall16[(size_t)p0 * 256 + idx];
  __syncthreads();

  // mode-probability head: relu(f_all@mp_w1^T+b1)
  for (int nt = 0; nt < 8; ++nt) {
    int n0 = nt * 16;
    v8f acc = bcast8(mp_b1[n0 + (lane & 15)]);
#pragma unroll
    for (int k0 = 0; k0 < 256; k0 += 32)
      acc = wmma16(frag_A(sF + k0, 256), frag_B(mpw1_16, 256, n0, k0), acc);
    store_D(sH1 + n0, 128, acc);
  }
  __syncthreads();
  for (int idx = lane; idx < 16 * 128; idx += 32) sH1[idx] = fmaxf(sH1[idx], 0.f);
  __syncthreads();
  if (lane < 16) {
    int p = lane;
    float lg[MODES]; float mx = -1e30f;
    for (int m = 0; m < MODES; ++m) {
      float a = mp_b2[m];
      for (int d = 0; d < 128; ++d) a += sH1[p * 128 + d] * mp_w2[m * 128 + d];
      lg[m] = a; mx = fmaxf(mx, a);
    }
    float sm = 0.f;
    for (int m = 0; m < MODES; ++m) { lg[m] = __expf(lg[m] - mx); sm += lg[m]; }
    float inv = 1.f / sm;
    float* pr = out + (size_t)NPED * MODES * PREDL * 2;
    for (int m = 0; m < MODES; ++m) pr[(size_t)(p0 + p) * MODES + m] = lg[m] * inv;
  }
  __syncthreads();

  // trajectory heads, per mode
  for (int m = 0; m < MODES; ++m) {
    const _Float16* W1 = tdw1_16 + (size_t)m * 128 * 256;
    const _Float16* W2 = tdw2_16 + (size_t)m * 128 * 128;
    const _Float16* W3 = tdw3_16 + (size_t)m * 32 * 128;
    // h1 = relu(f_all@W1^T + b1)
    for (int nt = 0; nt < 8; ++nt) {
      int n0 = nt * 16;
      v8f acc = bcast8(td_b1[m * 128 + n0 + (lane & 15)]);
#pragma unroll
      for (int k0 = 0; k0 < 256; k0 += 32)
        acc = wmma16(frag_A(sF + k0, 256), frag_B(W1, 256, n0, k0), acc);
      store_D(sH1 + n0, 128, acc);
    }
    __syncthreads();
    for (int idx = lane; idx < 16 * 128; idx += 32) sHh[idx] = (_Float16)fmaxf(sH1[idx], 0.f);
    __syncthreads();
    // h2 = relu(h1@W2^T + b2)
    for (int nt = 0; nt < 8; ++nt) {
      int n0 = nt * 16;
      v8f acc = bcast8(td_b2[m * 128 + n0 + (lane & 15)]);
#pragma unroll
      for (int k0 = 0; k0 < 128; k0 += 32)
        acc = wmma16(frag_A(sHh + k0, 128), frag_B(W2, 128, n0, k0), acc);
      store_D(sH1 + n0, 128, acc);
    }
    __syncthreads();
    for (int idx = lane; idx < 16 * 128; idx += 32) sHh[idx] = (_Float16)fmaxf(sH1[idx], 0.f);
    __syncthreads();
    // traj = h2@W3^T + b3 (24 cols padded to 32)
    for (int nt = 0; nt < 2; ++nt) {
      int n0 = nt * 16;
      int ncol = n0 + (lane & 15);
      v8f acc = bcast8((ncol < PREDL * 2) ? td_b3[m * PREDL * 2 + ncol] : 0.f);
#pragma unroll
      for (int k0 = 0; k0 < 128; k0 += 32)
        acc = wmma16(frag_A(sHh + k0, 128), frag_B(W3, 128, n0, k0), acc);
      store_D(sT + n0, 32, acc);
    }
    __syncthreads();
    // cumsum + last_pos -> predictions
    if (lane < 16) {
      int p = lane;
      float ax = obs[(size_t)(OBSL - 1) * NPED * 2 + (size_t)(p0 + p) * 2 + 0];
      float ay = obs[(size_t)(OBSL - 1) * NPED * 2 + (size_t)(p0 + p) * 2 + 1];
      float* o = out + (((size_t)(p0 + p) * MODES + m) * PREDL) * 2;
      for (int t = 0; t < PREDL; ++t) {
        ax += sT[p * 32 + 2 * t];
        ay += sT[p * 32 + 2 * t + 1];
        o[2 * t] = ax; o[2 * t + 1] = ay;
      }
    }
    __syncthreads();
  }
}

// ---------------- host launcher ----------------
extern "C" void kernel_launch(void* const* d_in, const int* in_sizes, int n_in,
                              void* d_out, int out_size, void* d_ws, size_t ws_size,
                              hipStream_t stream) {
  (void)in_sizes; (void)n_in; (void)out_size; (void)ws_size;
  const float* obs   = (const float*)d_in[0];
  const float* pe_w  = (const float*)d_in[1];
  const float* pe_b  = (const float*)d_in[2];
  const float* wih   = (const float*)d_in[3];
  const float* bih   = (const float*)d_in[4];
  const float* whh   = (const float*)d_in[5];
  const float* bhh   = (const float*)d_in[6];
  const float* rp_w1 = (const float*)d_in[7];
  const float* rp_b1 = (const float*)d_in[8];
  const float* rp_w2 = (const float*)d_in[9];
  const float* rp_b2 = (const float*)d_in[10];
  const float* wq    = (const float*)d_in[11];
  const float* bq    = (const float*)d_in[12];
  const float* wk    = (const float*)d_in[13];
  const float* bk    = (const float*)d_in[14];
  const float* wv    = (const float*)d_in[15];
  const float* bv    = (const float*)d_in[16];
  const float* wo    = (const float*)d_in[17];
  const float* bo    = (const float*)d_in[18];
  const float* op_w  = (const float*)d_in[19];
  const float* op_b  = (const float*)d_in[20];
  const float* mp_w1 = (const float*)d_in[21];
  const float* mp_b1 = (const float*)d_in[22];
  const float* mp_w2 = (const float*)d_in[23];
  const float* mp_b2 = (const float*)d_in[24];
  const float* td_w1 = (const float*)d_in[25];
  const float* td_b1 = (const float*)d_in[26];
  const float* td_w2 = (const float*)d_in[27];
  const float* td_b2 = (const float*)d_in[28];
  const float* td_w3 = (const float*)d_in[29];
  const float* td_b3 = (const float*)d_in[30];
  float* out = (float*)d_out;

  _Float16* wpool = (_Float16*)d_ws;
  size_t f16_bytes = F16_TOTAL * 2;
  float* nf32 = (float*)((char*)d_ws + ((f16_bytes + 255) & ~(size_t)255));
  float* q32  = nf32 + (size_t)NPED * 128;

  // weight conversion + fragment-major swizzle into workspace pool
  struct { const float* src; size_t off; int out_d; int in_d; } cv[] = {
    { wih,   O_WIH,  512, 64  }, { whh,   O_WHH,  512, 128 },
    { rp_w2, O_RPW2, 128, 64  }, { wk,    O_WK,   128, 128 },
    { wv,    O_WV,   128, 128 }, { wq,    O_WQ,   128, 128 },
    { wo,    O_WO,   128, 128 }, { op_w,  O_OPW,  128, 256 },
    { mp_w1, O_MPW1, 128, 256 }, { td_w1, O_TDW1, 5 * 128, 256 },
    { td_w2, O_TDW2, 5 * 128, 128 },
  };
  for (auto& c : cv) {
    int n = c.out_d * c.in_d;
    k_swz<<<(n + 255) / 256, 256, 0, stream>>>(c.src, wpool + c.off, c.out_d, c.in_d);
  }
  k_swz_tdw3<<<(MODES * 32 * 128 + 255) / 256, 256, 0, stream>>>(td_w3, wpool + O_TDW3);

  k_lstm<<<NPED / 16, 32, 0, stream>>>(obs, pe_w, pe_b, bih, bhh,
                                       wpool + O_WIH, wpool + O_WHH,
                                       nf32, wpool + O_FALL);

  k_qproj<<<NPED / 16, 32, 0, stream>>>(wpool + O_FALL, wpool + O_WQ, bq, q32);

  k_attn<<<NPED, 32, 0, stream>>>(obs, rp_w1, rp_b1, wpool + O_RPW2, rp_b2,
                                  wpool + O_WK, bk, wpool + O_WV, bv,
                                  q32, nf32, wpool + O_CTX);

  k_edge<<<NPED / 16, 32, 0, stream>>>(wpool + O_FALL, wpool + O_CTX,
                                       wpool + O_WO, bo, wpool + O_OPW, op_b,
                                       wpool + O_FALL);

  k_dec<<<NPED / 16, 32, 0, stream>>>(wpool + O_FALL, wpool + O_MPW1, mp_b1,
                                      mp_w2, mp_b2, wpool + O_TDW1, td_b1,
                                      wpool + O_TDW2, td_b2, wpool + O_TDW3, td_b3,
                                      obs, out);
}
